// TextGuidedAdapter_61426622268100
// MI455X (gfx1250) — matrix operand order repaired and checked by reading, not compiled, not run on hardware
//
#include <hip/hip_runtime.h>

// ---------------- constants (static shapes from the reference) ----------------
#define BSZ   2
#define DM    256
#define NHD   8
#define STOT  10200
#define N16   1920
#define LT    110
#define LTPAD 128
#define OFF1  7680      // level-1 start
// levels: (48,160) (24,80) (12,40) (6,20); offsets 0,7680,9600,10080

typedef __bf16 bf16_t;
typedef bf16_t bf16x16 __attribute__((ext_vector_type(16)));
typedef float  floatx8 __attribute__((ext_vector_type(8)));
typedef unsigned int u32x4 __attribute__((ext_vector_type(4)));

union Frag { bf16x16 v; u32x4 q[2]; };

__device__ __forceinline__ unsigned short f2bf(float x) {
  unsigned int u = __float_as_uint(x);
  unsigned int r = u + 0x7FFFu + ((u >> 16) & 1u);
  return (unsigned short)(r >> 16);
}

// ---------------- generic batched WMMA GEMM: C = alpha*(A @ W^T) + bias ------
// A: (M,K) bf16 row-major, lda elements. W: (N,K) bf16 row-major, ldw elements.
// C: (M,N) f32, ldc. One wave computes a 64(M)x16(N) tile with 4 accumulators:
// the B fragment is loaded once per K-step and reused by 4 WMMAs (2.5 vector
// loads per WMMA instead of 4). grid.z batches over bh = bh0+z with b=bh/8,
// h=bh%8 and separate b/h/z strides per operand, so the same kernel serves
// plain GEMMs, per-head Q*K^T score GEMMs and P*V GEMMs.
struct GemmArgs {
  const unsigned short* A;
  const unsigned short* W;
  const float* bias;
  float* C;
  int M, N, K, lda, ldw, ldc;
  float alpha;
  int bh0;
  long long sAb, sAh, sAz, sWb, sWh, sWz, sCb, sCh, sCz;
};

__global__ __launch_bounds__(32) void k_gemm(GemmArgs g) {
  const int lane = threadIdx.x;
  const int kh = lane >> 4;       // which 16-lane half
  const int ln = lane & 15;
  const int bm = blockIdx.x * 64;
  const int bn = blockIdx.y * 16;
  const int z  = blockIdx.z;
  const int bh = g.bh0 + z;
  const int b = bh >> 3, h = bh & 7;

  const unsigned short* A = g.A + (long long)b * g.sAb + (long long)h * g.sAh + (long long)z * g.sAz;
  const unsigned short* W = g.W + (long long)b * g.sWb + (long long)h * g.sWh + (long long)z * g.sWz;
  float* C = g.C + (long long)b * g.sCb + (long long)h * g.sCh + (long long)z * g.sCz;

  const int col = bn + ln;
  const unsigned short* wp = W + (long long)col * g.ldw;

  const unsigned short* ap[4];
#pragma unroll
  for (int t = 0; t < 4; ++t) {
    int row = bm + 16 * t + ln;                 // A frag rows: M = lane%16
    if (row > g.M - 1) row = g.M - 1;           // clamp for M edge
    ap[t] = A + (long long)row * g.lda;
  }

  floatx8 acc0 = {0.f, 0.f, 0.f, 0.f, 0.f, 0.f, 0.f, 0.f};
  floatx8 acc1 = acc0, acc2 = acc0, acc3 = acc0;

  for (int k0 = 0; k0 < g.K; k0 += 32) {
    // B 32x16 bf16 layout: e<8 -> K=8*kh+e ; e>=8 -> K=16+8*kh+(e-8), n=lane%16
    Frag fb;
    fb.q[0] = *(const u32x4*)(wp + k0 + 8 * kh);
    fb.q[1] = *(const u32x4*)(wp + k0 + 16 + 8 * kh);
    // A 16x32 bf16 layout: e<8 -> K=8*kh+e ; e>=8 -> K=16+8*kh+(e-8)
    Frag fa0, fa1, fa2, fa3;
    fa0.q[0] = *(const u32x4*)(ap[0] + k0 + 8 * kh);
    fa0.q[1] = *(const u32x4*)(ap[0] + k0 + 16 + 8 * kh);
    fa1.q[0] = *(const u32x4*)(ap[1] + k0 + 8 * kh);
    fa1.q[1] = *(const u32x4*)(ap[1] + k0 + 16 + 8 * kh);
    fa2.q[0] = *(const u32x4*)(ap[2] + k0 + 8 * kh);
    fa2.q[1] = *(const u32x4*)(ap[2] + k0 + 16 + 8 * kh);
    fa3.q[0] = *(const u32x4*)(ap[3] + k0 + 8 * kh);
    fa3.q[1] = *(const u32x4*)(ap[3] + k0 + 16 + 8 * kh);
    acc0 = __builtin_amdgcn_wmma_f32_16x16x32_bf16(false, fa0.v, false, fb.v, (short)0, acc0, false, false);
    acc1 = __builtin_amdgcn_wmma_f32_16x16x32_bf16(false, fa1.v, false, fb.v, (short)0, acc1, false, false);
    acc2 = __builtin_amdgcn_wmma_f32_16x16x32_bf16(false, fa2.v, false, fb.v, (short)0, acc2, false, false);
    acc3 = __builtin_amdgcn_wmma_f32_16x16x32_bf16(false, fa3.v, false, fb.v, (short)0, acc3, false, false);
  }

  const float bv = g.bias ? g.bias[col] : 0.f;
  // C: VGPR v -> M = v + 8*(lane>=16), N = lane%16
#pragma unroll
  for (int t = 0; t < 4; ++t) {
    const floatx8 acc = (t == 0) ? acc0 : (t == 1) ? acc1 : (t == 2) ? acc2 : acc3;
    const int mb = bm + 16 * t + 8 * kh;
#pragma unroll
    for (int v = 0; v < 8; ++v) {
      const int m = mb + v;
      if (m < g.M) C[(long long)m * g.ldc + col] = acc[v] * g.alpha + bv;
    }
  }
}

// ---------------- elementwise / conversion kernels ----------------------------
__global__ void k_cvt(unsigned short* d, const float* s, int n) {
  int i = blockIdx.x * 256 + threadIdx.x; if (i < n) d[i] = f2bf(s[i]);
}
__global__ void k_cvt_add(unsigned short* d, const float* a, const float* b, int n) {
  int i = blockIdx.x * 256 + threadIdx.x; if (i < n) d[i] = f2bf(a[i] + b[i]);
}
// slice of level-1 tokens out of (B,STOT,DM), optional add of second source
__global__ void k_cvt_slice(unsigned short* d, const float* src, const float* src2, int n) {
  int i = blockIdx.x * 256 + threadIdx.x; if (i >= n) return;
  int b = i / (N16 * DM);
  size_t j = (size_t)b * (STOT * DM) + (size_t)OFF1 * DM + (size_t)(i - b * (N16 * DM));
  float v = src[j]; if (src2) v += src2[j];
  d[i] = f2bf(v);
}
// convert (B*rows,DM) f32 into (B*padrows,DM) bf16, zero-filling pad rows
__global__ void k_cvt_pad(unsigned short* d, const float* s, int rows, int padrows, int n) {
  int i = blockIdx.x * 256 + threadIdx.x; if (i >= n) return;
  int b = i / (padrows * DM); int r = (i / DM) % padrows; int c = i % DM;
  d[i] = (r < rows) ? f2bf(s[((size_t)(b * rows + r)) * DM + c]) : (unsigned short)0;
}
// per-head transpose of V: (B,Lk,DM) f32 -> (B,8,32,LkPad) bf16, zero-padded
__global__ void k_vpt(unsigned short* d, const float* vp, int Lk, int LkPad, int n) {
  int i = blockIdx.x * 256 + threadIdx.x; if (i >= n) return;
  int k = i % LkPad; int c = (i / LkPad) % 32; int h = (i / (LkPad * 32)) % 8;
  int b = i / (LkPad * 32 * 8);
  d[i] = (k < Lk) ? f2bf(vp[((size_t)(b * Lk + k)) * DM + h * 32 + c]) : (unsigned short)0;
}
// hq = (src with level-1 replaced by f1+imgfeat_adapt) + pos, as bf16
__global__ void k_hq(unsigned short* d, const float* img, const float* pos,
                     const float* ifa, int n) {
  int i = blockIdx.x * 256 + threadIdx.x; if (i >= n) return;
  int c = i % DM; int t = (i / DM) % STOT; int b = i / (DM * STOT);
  float v = img[i] + pos[i];
  if (t >= OFF1 && t < OFF1 + N16) v += ifa[((size_t)(b * N16 + (t - OFF1))) * DM + c];
  d[i] = f2bf(v);
}
__global__ void k_add(float* d, const float* a, const float* b, int n) {
  int i = blockIdx.x * 256 + threadIdx.x; if (i < n) d[i] = a[i] + b[i];
}
__global__ void k_maskval(float* v, const unsigned char* m, int n) {
  int i = blockIdx.x * 256 + threadIdx.x; if (i < n && m[i / DM]) v[i] = 0.f;
}

// ---------------- softmax over keys (per row), writes zero-padded bf16 -------
__global__ __launch_bounds__(128) void k_softmax(const float* S, unsigned short* Sb,
    int Lq, int Lk, int LkPad, const unsigned char* kpm, int bh0) {
  int r = blockIdx.x * 4 + (threadIdx.x >> 5);
  if (r >= Lq) return;
  int lane = threadIdx.x & 31;
  size_t zoff = (size_t)blockIdx.z * Lq * LkPad;
  const float* row = S + zoff + (size_t)r * LkPad;
  unsigned short* orow = Sb + zoff + (size_t)r * LkPad;
  const unsigned char* mrow = kpm ? kpm + (size_t)((bh0 + blockIdx.z) >> 3) * Lk : (const unsigned char*)0;
  float mx = -1e30f;
  for (int k = lane; k < Lk; k += 32) {
    float v = row[k]; if (mrow && mrow[k]) v = -1e9f;
    mx = fmaxf(mx, v);
  }
  for (int o = 16; o; o >>= 1) mx = fmaxf(mx, __shfl_xor(mx, o));
  float sum = 0.f;
  for (int k = lane; k < Lk; k += 32) {
    float v = row[k]; if (mrow && mrow[k]) v = -1e9f;
    sum += expf(v - mx);
  }
  for (int o = 16; o; o >>= 1) sum += __shfl_xor(sum, o);
  float inv = 1.f / sum;
  for (int k = lane; k < LkPad; k += 32) {
    float p = 0.f;
    if (k < Lk) {
      float v = row[k]; if (mrow && mrow[k]) v = -1e9f;
      p = expf(v - mx) * inv;
    }
    orow[k] = f2bf(p);
  }
}

// ---------------- MS-deformable-attention sampling ----------------------------
// one wave per (b,q,h); lane = channel. Does aw softmax + 16 bilinear samples.
__global__ __launch_bounds__(256) void k_msda(const float* value, const float* off,
    const float* awl, const float* vr, float* outp, int total) {
  int wid = (blockIdx.x * 256 + threadIdx.x) >> 5;
  int lane = threadIdx.x & 31;
  if (wid >= total) return;
  int h = wid % 8; int q = (wid / 8) % STOT; int b = wid / (8 * STOT);
  const int HH[4] = {48, 24, 12, 6}, WW[4] = {160, 80, 40, 20}, OF[4] = {0, 7680, 9600, 10080};
  int ql = (q < 7680) ? 0 : (q < 9600) ? 1 : (q < 10080) ? 2 : 3;
  int p = q - OF[ql];
  int iy = p / WW[ql], ix = p % WW[ql];
  float rx0 = (ix + 0.5f) / (vr[(b * 4 + ql) * 2 + 0] * (float)WW[ql]);
  float ry0 = (iy + 0.5f) / (vr[(b * 4 + ql) * 2 + 1] * (float)HH[ql]);
  const float* aro = awl + ((size_t)(b * STOT + q)) * 128 + h * 16;
  float w[16]; float mx = -1e30f;
  for (int i = 0; i < 16; i++) { w[i] = aro[i]; mx = fmaxf(mx, w[i]); }
  float sm = 0.f;
  for (int i = 0; i < 16; i++) { w[i] = expf(w[i] - mx); sm += w[i]; }
  float inv = 1.f / sm;
  const float* orow = off + ((size_t)(b * STOT + q)) * 256 + h * 32;
  float acc = 0.f;
  for (int l = 0; l < 4; l++) {
    int W_ = WW[l], H_ = HH[l];
    float refx = rx0 * vr[(b * 4 + l) * 2 + 0];
    float refy = ry0 * vr[(b * 4 + l) * 2 + 1];
    const float* vbase = value + ((size_t)(b * STOT + OF[l])) * 256 + h * 32 + lane;
    for (int pt = 0; pt < 4; pt++) {
      float ox = orow[(l * 4 + pt) * 2 + 0];
      float oy = orow[(l * 4 + pt) * 2 + 1];
      float x = (refx + ox / (float)W_) * (float)W_ - 0.5f;
      float y = (refy + oy / (float)H_) * (float)H_ - 0.5f;
      float xf = floorf(x), yf = floorf(y);
      int x0 = (int)xf, y0 = (int)yf;
      float wx = x - xf, wy = y - yf;
      float aww = w[l * 4 + pt] * inv;
      float s = 0.f;
      if (x0     >= 0 && x0     < W_ && y0     >= 0 && y0     < H_) s += (1 - wx) * (1 - wy) * vbase[((size_t)(y0 * W_ + x0)) * 256];
      if (x0 + 1 >= 0 && x0 + 1 < W_ && y0     >= 0 && y0     < H_) s += wx * (1 - wy)       * vbase[((size_t)(y0 * W_ + x0 + 1)) * 256];
      if (x0     >= 0 && x0     < W_ && y0 + 1 >= 0 && y0 + 1 < H_) s += (1 - wx) * wy       * vbase[((size_t)((y0 + 1) * W_ + x0)) * 256];
      if (x0 + 1 >= 0 && x0 + 1 < W_ && y0 + 1 >= 0 && y0 + 1 < H_) s += wx * wy             * vbase[((size_t)((y0 + 1) * W_ + x0 + 1)) * 256];
      acc += aww * s;
    }
  }
  outp[((size_t)(b * STOT + q)) * 256 + h * 32 + lane] = acc;
}

// ---------------- verify score: cosine(l2norm) -> gaussian gate ---------------
__global__ __launch_bounds__(128) void k_verify(const float* oe, const float* ae,
    const float* scale, const float* sigma, float* vs, int rows) {
  int r = blockIdx.x * 4 + (threadIdx.x >> 5);
  if (r >= rows) return;
  int lane = threadIdx.x & 31;
  const float* o = oe + (size_t)r * DM;
  const float* a = ae + (size_t)r * DM;
  float d = 0.f, no = 0.f, na = 0.f;
  for (int c = lane; c < DM; c += 32) { float x = o[c], y = a[c]; d += x * y; no += x * x; na += y * y; }
  for (int off = 16; off; off >>= 1) {
    d += __shfl_xor(d, off); no += __shfl_xor(no, off); na += __shfl_xor(na, off);
  }
  if (lane == 0) {
    float nn = fmaxf(sqrtf(no), 1e-12f) * fmaxf(sqrtf(na), 1e-12f);
    float u = 1.0f - d / nn;
    float sg = sigma[0];
    vs[r] = scale[0] * expf(-(u * u) / (2.f * sg * sg));
  }
}

// 2x bilinear upsample (24,80)->(48,160), half-pixel centers, edge clamp
__global__ void k_resize(float* dst, const float* src, int n) {
  int i = blockIdx.x * 256 + threadIdx.x; if (i >= n) return;
  int b = i / (48 * 160); int r = i % (48 * 160); int oy = r / 160, ox = r % 160;
  float sy = (oy + 0.5f) * 0.5f - 0.5f;
  float sx = (ox + 0.5f) * 0.5f - 0.5f;
  float yf = floorf(sy), xf = floorf(sx);
  int y0 = (int)yf, x0 = (int)xf;
  float wy = sy - yf, wx = sx - xf;
  int y0c = min(max(y0, 0), 23), y1c = min(max(y0 + 1, 0), 23);
  int x0c = min(max(x0, 0), 79), x1c = min(max(x0 + 1, 0), 79);
  const float* s = src + b * 1920;
  dst[i] = (1 - wy) * ((1 - wx) * s[y0c * 80 + x0c] + wx * s[y0c * 80 + x1c])
         + wy       * ((1 - wx) * s[y1c * 80 + x0c] + wx * s[y1c * 80 + x1c]);
}

// 3x3 stride2 pad1 maxpool
__global__ void k_maxpool(float* dst, const float* src, int H, int W, int Ho, int Wo, int n) {
  int i = blockIdx.x * 256 + threadIdx.x; if (i >= n) return;
  int b = i / (Ho * Wo); int r = i % (Ho * Wo); int oy = r / Wo, ox = r % Wo;
  const float* s = src + b * H * W;
  float m = -3.0e38f;
  for (int dy = -1; dy <= 1; dy++) {
    int y = oy * 2 + dy; if (y < 0 || y >= H) continue;
    for (int dx = -1; dx <= 1; dx++) {
      int x = ox * 2 + dx; if (x < 0 || x >= W) continue;
      m = fmaxf(m, s[y * W + x]);
    }
  }
  dst[i] = m;
}

// ---------------- fused LN + gate + concat outputs ----------------------------
__global__ __launch_bounds__(128) void k_out_img(const float* img, const float* ctx,
    const float* g1, const float* b1, const float* g2, const float* b2,
    const float* vs8, const float* vs16, const float* vs32, const float* vs64,
    float* out, int rows) {
  int r = blockIdx.x * 4 + (threadIdx.x >> 5);
  if (r >= rows) return;
  int lane = threadIdx.x & 31;
  int b = r / STOT, t = r % STOT;
  const float* x = img + (size_t)r * DM;
  const float* y = ctx + (size_t)r * DM;
  float s1 = 0.f, q1 = 0.f, s2 = 0.f, q2 = 0.f;
  for (int c = lane; c < DM; c += 32) {
    float a = x[c]; s1 += a; q1 += a * a;
    float d = y[c]; s2 += d; q2 += d * d;
  }
  for (int o = 16; o; o >>= 1) {
    s1 += __shfl_xor(s1, o); q1 += __shfl_xor(q1, o);
    s2 += __shfl_xor(s2, o); q2 += __shfl_xor(q2, o);
  }
  float m1 = s1 / 256.f, v1 = q1 / 256.f - m1 * m1, r1 = rsqrtf(v1 + 1e-5f);
  float m2 = s2 / 256.f, v2 = q2 / 256.f - m2 * m2, r2 = rsqrtf(v2 + 1e-5f);
  float vv;
  if (t < 7680)       vv = vs8[b * 7680 + t];
  else if (t < 9600)  vv = vs16[b * 1920 + t - 7680];
  else if (t < 10080) vv = vs32[b * 480 + t - 9600];
  else                vv = vs64[b * 120 + t - 10080];
  float* orow = out + (size_t)r * 512;
  for (int c = lane; c < DM; c += 32) {
    float a = x[c], d = y[c];
    float l1 = (a - m1) * r1 * g1[c] + b1[c];
    float l2 = (d - m2) * r2 * g2[c] + b2[c];
    orow[c] = a;
    orow[256 + c] = (l1 + l2) * vv;
  }
}

__global__ __launch_bounds__(128) void k_out_depth(const float* dpe, const float* tcd,
    const float* g1, const float* b1, const float* g2, const float* b2,
    const float* vsc, float* out, int rows) {
  int r = blockIdx.x * 4 + (threadIdx.x >> 5);
  if (r >= rows) return;
  int lane = threadIdx.x & 31;
  const float* x = dpe + (size_t)r * DM;
  const float* y = tcd + (size_t)r * DM;
  float s1 = 0.f, q1 = 0.f, s2 = 0.f, q2 = 0.f;
  for (int c = lane; c < DM; c += 32) {
    float a = x[c]; s1 += a; q1 += a * a;
    float d = y[c]; s2 += d; q2 += d * d;
  }
  for (int o = 16; o; o >>= 1) {
    s1 += __shfl_xor(s1, o); q1 += __shfl_xor(q1, o);
    s2 += __shfl_xor(s2, o); q2 += __shfl_xor(q2, o);
  }
  float m1 = s1 / 256.f, v1 = q1 / 256.f - m1 * m1, r1 = rsqrtf(v1 + 1e-5f);
  float m2 = s2 / 256.f, v2 = q2 / 256.f - m2 * m2, r2 = rsqrtf(v2 + 1e-5f);
  float vv = vsc[r];
  float* orow = out + (size_t)r * 512;
  for (int c = lane; c < DM; c += 32) {
    float a = x[c], d = y[c];
    float l1 = (a - m1) * r1 * g1[c] + b1[c];
    float l2 = (d - m2) * r2 * g2[c] + b2[c];
    orow[c] = a;
    orow[256 + c] = (l1 + l2) * vv;
  }
}

// ---------------- host side ----------------------------------------------------
static inline int cdiv(int a, int b) { return (a + b - 1) / b; }

static void gemm(hipStream_t st, const unsigned short* A, const unsigned short* W,
                 const float* bias, float* C, int M, int N, int K,
                 int lda, int ldw, int ldc, float alpha = 1.f, int nbh = 1, int bh0 = 0,
                 long long sAb = 0, long long sAh = 0, long long sAz = 0,
                 long long sWb = 0, long long sWh = 0, long long sWz = 0,
                 long long sCb = 0, long long sCh = 0, long long sCz = 0) {
  GemmArgs g{A, W, bias, C, M, N, K, lda, ldw, ldc, alpha, bh0,
             sAb, sAh, sAz, sWb, sWh, sWz, sCb, sCh, sCz};
  dim3 grid(cdiv(M, 64), N / 16, nbh);
  k_gemm<<<grid, dim3(32), 0, st>>>(g);
}

extern "C" void kernel_launch(void* const* d_in, const int* in_sizes, int n_in,
                              void* d_out, int out_size, void* d_ws, size_t ws_size,
                              hipStream_t stream) {
  (void)in_sizes; (void)n_in; (void)out_size; (void)ws_size;
  // ---- inputs (setup_inputs dict order, params flattened in insertion order)
  const float* img  = (const float*)d_in[0];
  const unsigned char* masks = (const unsigned char*)d_in[1];
  const float* pos  = (const float*)d_in[2];
  const float* vr   = (const float*)d_in[5];
  const float* wordf = (const float*)d_in[6];
  const unsigned char* wkpm = (const unsigned char*)d_in[7];
  const float* dpe  = (const float*)d_in[8];
  const unsigned char* mdep = (const unsigned char*)d_in[9];
  const float* wpos = (const float*)d_in[10];
  const float* w_i2t_in  = (const float*)d_in[11]; const float* b_i2t_in  = (const float*)d_in[12];
  const float* w_i2t_out = (const float*)d_in[13]; const float* b_i2t_out = (const float*)d_in[14];
  const float* w_d2t_in  = (const float*)d_in[15]; const float* b_d2t_in  = (const float*)d_in[16];
  const float* w_d2t_out = (const float*)d_in[17]; const float* b_d2t_out = (const float*)d_in[18];
  const float* w_d2d_in  = (const float*)d_in[19]; const float* b_d2d_in  = (const float*)d_in[20];
  const float* w_d2d_out = (const float*)d_in[21]; const float* b_d2d_out = (const float*)d_in[22];
  const float* adapt_w = (const float*)d_in[23]; const float* adapt_b = (const float*)d_in[24];
  const float* orig_w  = (const float*)d_in[25]; const float* orig_b  = (const float*)d_in[26];
  const float* tf_scale = (const float*)d_in[27]; const float* tf_sigma = (const float*)d_in[28];
  const float* off_w  = (const float*)d_in[29]; const float* off_b  = (const float*)d_in[30];
  const float* aw_w   = (const float*)d_in[31]; const float* aw_b   = (const float*)d_in[32];
  const float* val_w  = (const float*)d_in[33]; const float* val_b  = (const float*)d_in[34];
  const float* mout_w = (const float*)d_in[35]; const float* mout_b = (const float*)d_in[36];
  const float* ln_img_g = (const float*)d_in[37]; const float* ln_img_b = (const float*)d_in[38];
  const float* ln_tci_g = (const float*)d_in[39]; const float* ln_tci_b = (const float*)d_in[40];
  const float* ln_dep_g = (const float*)d_in[41]; const float* ln_dep_b = (const float*)d_in[42];
  const float* ln_tcd_g = (const float*)d_in[43]; const float* ln_tcd_b = (const float*)d_in[44];

  float* out1 = (float*)d_out;
  float* out2 = out1 + (size_t)BSZ * STOT * 512;

  // ---- workspace allocator
  char* base = (char*)d_ws;
  size_t woff = 0;
  auto A16 = [&](size_t e) { woff = (woff + 255) & ~(size_t)255; unsigned short* p = (unsigned short*)(base + woff); woff += e * 2; return p; };
  auto A32 = [&](size_t e) { woff = (woff + 255) & ~(size_t)255; float* p = (float*)(base + woff); woff += e * 4; return p; };

  const size_t nBND = (size_t)BSZ * N16 * DM;   // 983040
  const size_t nBSD = (size_t)BSZ * STOT * DM;  // 5222400
  const size_t nBLD = (size_t)BSZ * LT * DM;    // 56320

  // bf16 weights
  unsigned short* wb_i2t_in  = A16(768 * 256);
  unsigned short* wb_i2t_out = A16(256 * 256);
  unsigned short* wb_d2t_in  = A16(768 * 256);
  unsigned short* wb_d2t_out = A16(256 * 256);
  unsigned short* wb_d2d_in  = A16(768 * 256);
  unsigned short* wb_d2d_out = A16(256 * 256);
  unsigned short* wb_adapt = A16(256 * 256);
  unsigned short* wb_orig  = A16(256 * 256);
  unsigned short* wb_off   = A16(256 * 256);
  unsigned short* wb_aw    = A16(128 * 256);
  unsigned short* wb_val   = A16(256 * 256);
  unsigned short* wb_mout  = A16(256 * 256);
  // bf16 activations
  unsigned short* qin_bf = A16(nBND);
  unsigned short* kin_bf = A16(nBLD);
  unsigned short* vin_bf = A16(nBLD);
  unsigned short* qp_bf  = A16(nBND);
  unsigned short* kp_bf  = A16((size_t)BSZ * LTPAD * DM);
  unsigned short* vpt_bf = A16((size_t)BSZ * 8 * 32 * LTPAD);
  unsigned short* sb_pool = A16((size_t)16 * N16 * LTPAD);
  unsigned short* attno_bf = A16(nBND);
  unsigned short* ifa_bf = A16(nBND);
  unsigned short* f1_bf  = A16(nBND);
  unsigned short* hq_bf  = A16(nBSD);
  unsigned short* img_bf = A16(nBSD);
  unsigned short* samp_bf = A16(nBSD);
  unsigned short* dpe_bf = A16(nBND);
  unsigned short* qd_bf  = A16(nBND);
  unsigned short* qp3_bf = A16(nBND);
  unsigned short* kp3_bf = A16(nBND);
  unsigned short* vpt3_bf = A16((size_t)BSZ * 8 * 32 * N16);
  unsigned short* sb2 = A16((size_t)2 * N16 * N16);
  // f32 scratch
  float* qp1f = A32(nBND);
  float* kp1f = A32(nBLD);
  float* vp1f = A32(nBLD);
  float* s_pool = A32((size_t)16 * N16 * LTPAD);
  float* attno = A32(nBND);
  float* ifa = A32(nBND);
  float* ae = A32(nBND);
  float* oe = A32(nBND);
  float* vs16b = A32((size_t)BSZ * 1920);
  float* vs8b  = A32((size_t)BSZ * 7680);
  float* vs32b = A32((size_t)BSZ * 480);
  float* vs64b = A32((size_t)BSZ * 120);
  float* valuef = A32(nBSD);
  float* offf = A32(nBSD);
  float* awlf = A32((size_t)BSZ * STOT * 128);
  float* sampf = A32(nBSD);
  float* ctxf = A32(nBSD);
  float* daf = A32(nBND);
  float* qdf = A32(nBND);
  float* qp3f = A32(nBND);
  float* kp3f = A32(nBND);
  float* vp3f = A32(nBND);
  float* s2 = A32((size_t)2 * N16 * N16);
  float* tcdf = A32(nBND);

  const float ATT_SCALE = 0.17677669529663687f; // 1/sqrt(32)
  auto EL = [&](int n) { return dim3(cdiv(n, 256)); };

  // ---- 0) weights to bf16
  k_cvt<<<EL(768 * 256), 256, 0, stream>>>(wb_i2t_in, w_i2t_in, 768 * 256);
  k_cvt<<<EL(256 * 256), 256, 0, stream>>>(wb_i2t_out, w_i2t_out, 256 * 256);
  k_cvt<<<EL(768 * 256), 256, 0, stream>>>(wb_d2t_in, w_d2t_in, 768 * 256);
  k_cvt<<<EL(256 * 256), 256, 0, stream>>>(wb_d2t_out, w_d2t_out, 256 * 256);
  k_cvt<<<EL(768 * 256), 256, 0, stream>>>(wb_d2d_in, w_d2d_in, 768 * 256);
  k_cvt<<<EL(256 * 256), 256, 0, stream>>>(wb_d2d_out, w_d2d_out, 256 * 256);
  k_cvt<<<EL(256 * 256), 256, 0, stream>>>(wb_adapt, adapt_w, 256 * 256);
  k_cvt<<<EL(256 * 256), 256, 0, stream>>>(wb_orig, orig_w, 256 * 256);
  k_cvt<<<EL(256 * 256), 256, 0, stream>>>(wb_off, off_w, 256 * 256);
  k_cvt<<<EL(128 * 256), 256, 0, stream>>>(wb_aw, aw_w, 128 * 256);
  k_cvt<<<EL(256 * 256), 256, 0, stream>>>(wb_val, val_w, 256 * 256);
  k_cvt<<<EL(256 * 256), 256, 0, stream>>>(wb_mout, mout_w, 256 * 256);

  // ---- 1) img2text MHA: q=f1+p1, k=word+wordpos, v=word
  k_cvt_slice<<<EL((int)nBND), 256, 0, stream>>>(qin_bf, img, pos, (int)nBND);
  k_cvt_add<<<EL((int)nBLD), 256, 0, stream>>>(kin_bf, wordf, wpos, (int)nBLD);
  k_cvt<<<EL((int)nBLD), 256, 0, stream>>>(vin_bf, wordf, (int)nBLD);
  gemm(stream, qin_bf, wb_i2t_in,            b_i2t_in,       qp1f, BSZ * N16, 256, 256, 256, 256, 256);
  gemm(stream, kin_bf, wb_i2t_in + 256 * 256, b_i2t_in + 256, kp1f, BSZ * LT, 256, 256, 256, 256, 256);
  gemm(stream, vin_bf, wb_i2t_in + 512 * 256, b_i2t_in + 512, vp1f, BSZ * LT, 256, 256, 256, 256, 256);
  k_cvt<<<EL((int)nBND), 256, 0, stream>>>(qp_bf, qp1f, (int)nBND);
  k_cvt_pad<<<EL(BSZ * LTPAD * DM), 256, 0, stream>>>(kp_bf, kp1f, LT, LTPAD, BSZ * LTPAD * DM);
  k_vpt<<<EL(BSZ * 8 * 32 * LTPAD), 256, 0, stream>>>(vpt_bf, vp1f, LT, LTPAD, BSZ * 8 * 32 * LTPAD);
  // scores (per b,h): N16 x LTPAD, K=32
  gemm(stream, qp_bf, kp_bf, nullptr, s_pool, N16, LTPAD, 32, 256, 256, LTPAD, ATT_SCALE, 16, 0,
       (long long)N16 * 256, 32, 0, (long long)LTPAD * 256, 32, 0, 0, 0, (long long)N16 * LTPAD);
  k_softmax<<<dim3(N16 / 4, 1, 16), 128, 0, stream>>>(s_pool, sb_pool, N16, LT, LTPAD, wkpm, 0);
  // P @ V: N16 x 32, K=LTPAD
  gemm(stream, sb_pool, vpt_bf, nullptr, attno, N16, 32, LTPAD, LTPAD, LTPAD, 256, 1.f, 16, 0,
       0, 0, (long long)N16 * LTPAD, (long long)8 * 32 * LTPAD, (long long)32 * LTPAD, 0,
       (long long)N16 * 256, 32, 0);
  k_cvt<<<EL((int)nBND), 256, 0, stream>>>(attno_bf, attno, (int)nBND);
  gemm(stream, attno_bf, wb_i2t_out, b_i2t_out, ifa, BSZ * N16, 256, 256, 256, 256, 256);

  // ---- 2) verify gate
  k_cvt<<<EL((int)nBND), 256, 0, stream>>>(ifa_bf, ifa, (int)nBND);
  gemm(stream, ifa_bf, wb_adapt, adapt_b, ae, BSZ * N16, 256, 256, 256, 256, 256);
  k_cvt_slice<<<EL((int)nBND), 256, 0, stream>>>(f1_bf, img, nullptr, (int)nBND);
  gemm(stream, f1_bf, wb_orig, orig_b, oe, BSZ * N16, 256, 256, 256, 256, 256);
  k_verify<<<dim3(BSZ * N16 / 4), 128, 0, stream>>>(oe, ae, tf_scale, tf_sigma, vs16b, BSZ * N16);
  k_resize<<<EL(BSZ * 7680), 256, 0, stream>>>(vs8b, vs16b, BSZ * 7680);
  k_maxpool<<<EL(BSZ * 480), 256, 0, stream>>>(vs32b, vs16b, 24, 80, 12, 40, BSZ * 480);
  k_maxpool<<<EL(BSZ * 120), 256, 0, stream>>>(vs64b, vs32b, 12, 40, 6, 20, BSZ * 120);

  // ---- 3) MS-deformable attention
  k_hq<<<EL((int)nBSD), 256, 0, stream>>>(hq_bf, img, pos, ifa, (int)nBSD);
  k_cvt<<<EL((int)nBSD), 256, 0, stream>>>(img_bf, img, (int)nBSD);
  gemm(stream, img_bf, wb_val, val_b, valuef, BSZ * STOT, 256, 256, 256, 256, 256);
  k_maskval<<<EL((int)nBSD), 256, 0, stream>>>(valuef, masks, (int)nBSD);
  gemm(stream, hq_bf, wb_off, off_b, offf, BSZ * STOT, 256, 256, 256, 256, 256);
  gemm(stream, hq_bf, wb_aw, aw_b, awlf, BSZ * STOT, 128, 256, 256, 256, 128);
  {
    int total = BSZ * STOT * 8;
    k_msda<<<dim3(cdiv(total, 8)), 256, 0, stream>>>(valuef, offf, awlf, vr, sampf, total);
  }
  k_cvt<<<EL((int)nBSD), 256, 0, stream>>>(samp_bf, sampf, (int)nBSD);
  gemm(stream, samp_bf, wb_mout, mout_b, ctxf, BSZ * STOT, 256, 256, 256, 256, 256);

  // ---- 4) image output: concat(img, (LN(img)+LN(ctx))*verify)
  k_out_img<<<dim3(BSZ * STOT / 4), 128, 0, stream>>>(img, ctxf, ln_img_g, ln_img_b,
      ln_tci_g, ln_tci_b, vs8b, vs16b, vs32b, vs64b, out1, BSZ * STOT);

  // ---- 5a) d2t MHA: q=depth_pos_embed, k=word+wordpos, v=word
  k_cvt<<<EL((int)nBND), 256, 0, stream>>>(dpe_bf, dpe, (int)nBND);
  gemm(stream, dpe_bf, wb_d2t_in,            b_d2t_in,       qp1f, BSZ * N16, 256, 256, 256, 256, 256);
  gemm(stream, kin_bf, wb_d2t_in + 256 * 256, b_d2t_in + 256, kp1f, BSZ * LT, 256, 256, 256, 256, 256);
  gemm(stream, vin_bf, wb_d2t_in + 512 * 256, b_d2t_in + 512, vp1f, BSZ * LT, 256, 256, 256, 256, 256);
  k_cvt<<<EL((int)nBND), 256, 0, stream>>>(qp_bf, qp1f, (int)nBND);
  k_cvt_pad<<<EL(BSZ * LTPAD * DM), 256, 0, stream>>>(kp_bf, kp1f, LT, LTPAD, BSZ * LTPAD * DM);
  k_vpt<<<EL(BSZ * 8 * 32 * LTPAD), 256, 0, stream>>>(vpt_bf, vp1f, LT, LTPAD, BSZ * 8 * 32 * LTPAD);
  gemm(stream, qp_bf, kp_bf, nullptr, s_pool, N16, LTPAD, 32, 256, 256, LTPAD, ATT_SCALE, 16, 0,
       (long long)N16 * 256, 32, 0, (long long)LTPAD * 256, 32, 0, 0, 0, (long long)N16 * LTPAD);
  k_softmax<<<dim3(N16 / 4, 1, 16), 128, 0, stream>>>(s_pool, sb_pool, N16, LT, LTPAD, wkpm, 0);
  gemm(stream, sb_pool, vpt_bf, nullptr, attno, N16, 32, LTPAD, LTPAD, LTPAD, 256, 1.f, 16, 0,
       0, 0, (long long)N16 * LTPAD, (long long)8 * 32 * LTPAD, (long long)32 * LTPAD, 0,
       (long long)N16 * 256, 32, 0);
  k_cvt<<<EL((int)nBND), 256, 0, stream>>>(attno_bf, attno, (int)nBND);
  gemm(stream, attno_bf, wb_d2t_out, b_d2t_out, daf, BSZ * N16, 256, 256, 256, 256, 256);

  // ---- 5b) d2d self-attention on qd = dpe + depth_adapt, v = dpe
  k_add<<<EL((int)nBND), 256, 0, stream>>>(qdf, dpe, daf, (int)nBND);
  k_cvt<<<EL((int)nBND), 256, 0, stream>>>(qd_bf, qdf, (int)nBND);
  gemm(stream, qd_bf,  wb_d2d_in,            b_d2d_in,       qp3f, BSZ * N16, 256, 256, 256, 256, 256);
  gemm(stream, qd_bf,  wb_d2d_in + 256 * 256, b_d2d_in + 256, kp3f, BSZ * N16, 256, 256, 256, 256, 256);
  gemm(stream, dpe_bf, wb_d2d_in + 512 * 256, b_d2d_in + 512, vp3f, BSZ * N16, 256, 256, 256, 256, 256);
  k_cvt<<<EL((int)nBND), 256, 0, stream>>>(qp3_bf, qp3f, (int)nBND);
  k_cvt<<<EL((int)nBND), 256, 0, stream>>>(kp3_bf, kp3f, (int)nBND);
  k_vpt<<<EL(BSZ * 8 * 32 * N16), 256, 0, stream>>>(vpt3_bf, vp3f, N16, N16, BSZ * 8 * 32 * N16);
  for (int c = 0; c < 8; ++c) {   // 16 bh, chunks of 2 to bound scratch
    int bh0 = c * 2;
    gemm(stream, qp3_bf, kp3_bf, nullptr, s2, N16, N16, 32, 256, 256, N16, ATT_SCALE, 2, bh0,
         (long long)N16 * 256, 32, 0, (long long)N16 * 256, 32, 0, 0, 0, (long long)N16 * N16);
    k_softmax<<<dim3(N16 / 4, 1, 2), 128, 0, stream>>>(s2, sb2, N16, N16, N16, mdep, bh0);
    gemm(stream, sb2, vpt3_bf, nullptr, attno, N16, 32, N16, N16, N16, 256, 1.f, 2, bh0,
         0, 0, (long long)N16 * N16, (long long)8 * 32 * N16, (long long)32 * N16, 0,
         (long long)N16 * 256, 32, 0);
  }
  k_cvt<<<EL((int)nBND), 256, 0, stream>>>(attno_bf, attno, (int)nBND);
  gemm(stream, attno_bf, wb_d2d_out, b_d2d_out, tcdf, BSZ * N16, 256, 256, 256, 256, 256);

  // ---- 6) depth output
  k_out_depth<<<dim3(BSZ * N16 / 4), 128, 0, stream>>>(dpe, tcdf, ln_dep_g, ln_dep_b,
      ln_tcd_g, ln_tcd_b, vs16b, out2, BSZ * N16);
}
